// SE3PointKernel_12824772346231
// MI455X (gfx1250) — matrix operand order repaired and checked by reading, not compiled, not run on hardware
//
#include <hip/hip_runtime.h>

typedef __attribute__((ext_vector_type(16))) _Float16 v16h;
typedef __attribute__((ext_vector_type(8)))  float    v8f;

#define NPAIR_TOTAL 65536   // N*M = 256*256
#define HIDDEN      64
#define NWEIGHT     384
#define PAIRS       32      // pairs per workgroup
#define H_STRIDE    72      // padded f16 row stride for h (bank spread)
#define W_STRIDE    388     // padded f32 row stride for weights (bank spread)

// ---------------------------------------------------------------------------
// Pre-pass: convert W2 (64x384 f32, row-major) to f16 in workspace. 48 KB,
// stays resident in L2 for the whole main kernel.
// ---------------------------------------------------------------------------
__global__ __launch_bounds__(256) void w2_to_f16(const float* __restrict__ W2,
                                                 _Float16* __restrict__ W2h) {
    int i = blockIdx.x * blockDim.x + threadIdx.x;
    if (i < HIDDEN * NWEIGHT) W2h[i] = (_Float16)W2[i];
}

// ---------------------------------------------------------------------------
// Fused SE3 point kernel: MLP (WMMA f16) + spherical-harmonic tensor product,
// 32 pairs per block, 8 waves.
// ---------------------------------------------------------------------------
__global__ __launch_bounds__(256) void se3_fused(
    const float* __restrict__ dmat,
    const float* __restrict__ Q00, const float* __restrict__ Q01,
    const float* __restrict__ Q10, const float* __restrict__ Q11,
    const float* __restrict__ W1,  const float* __restrict__ b1,
    const _Float16* __restrict__ W2h, const float* __restrict__ b2,
    float* __restrict__ out)
{
    __shared__ _Float16 h_lds[PAIRS][H_STRIDE];   // h activations, f16
    __shared__ float    w_lds[PAIRS][W_STRIDE];   // per-pair 384 weights
    __shared__ float    qy_lds[PAIRS][35];        // per-pair QY contractions

    const int tid      = threadIdx.x;
    const int lane     = tid & 31;
    const int wave     = tid >> 5;
    const int pairBase = blockIdx.x * PAIRS;

    // ---- Phase 1: h[pair][k] = relu(r*W1[k] + b1[k]); each thread does 8 k's
    {
        const int hp = tid >> 3;            // pair 0..31
        const int kb = (tid & 7) * 8;       // k base
        const float* dp = dmat + (size_t)(pairBase + hp) * 3;
        float x = dp[0], y = dp[1], z = dp[2];
        float r = sqrtf(fmaxf(x*x + y*y + z*z, 1e-12f));
        #pragma unroll
        for (int kk = 0; kk < 8; ++kk) {
            int k = kb + kk;
            float hv = fmaxf(r * W1[k] + b1[k], 0.0f);
            h_lds[hp][k] = (_Float16)hv;
        }
    }

    // ---- Phase 0: spherical harmonics + QY contractions (wave 0, lane=pair)
    if (wave == 0) {
        const float* dp = dmat + (size_t)(pairBase + lane) * 3;
        float x = dp[0], y = dp[1], z = dp[2];
        float r = sqrtf(fmaxf(x*x + y*y + z*z, 1e-12f));
        float xi = x / r, yi = y / r, zi = z / r;
        const float c1  = 0.48860251190291992f;
        const float c2a = 1.09254843059207907f;
        const float c2b = 0.31539156525252005f;
        const float c2c = 0.54627421529603953f;
        float Y[9];
        Y[0] = 0.28209479177387814f;
        Y[1] = c1 * yi;  Y[2] = c1 * zi;  Y[3] = c1 * xi;
        Y[4] = c2a * xi * yi;
        Y[5] = c2a * yi * zi;
        Y[6] = c2b * (3.0f * zi * zi - 1.0f);
        Y[7] = c2a * xi * zi;
        Y[8] = c2c * (xi * xi - yi * yi);

        float* q = qy_lds[lane];
        // block (l_out=0,l_in=0): J=0
        q[0] = Q00[0] * Y[0];
        // block (0,1): J=1, Q01 is (1,3,3)
        #pragma unroll
        for (int j = 0; j < 3; ++j)
            q[1 + j] = Q01[j*3+0]*Y[1] + Q01[j*3+1]*Y[2] + Q01[j*3+2]*Y[3];
        // block (1,0): J=1, Q10 is (3,1,3)
        #pragma unroll
        for (int i = 0; i < 3; ++i)
            q[4 + i] = Q10[i*3+0]*Y[1] + Q10[i*3+1]*Y[2] + Q10[i*3+2]*Y[3];
        // block (1,1): Js={0,1,2}, Q11 is (3,3,9) -> r slices [0],[1..3],[4..8]
        #pragma unroll
        for (int ij = 0; ij < 9; ++ij) {
            const float* qq = Q11 + ij * 9;
            q[7  + ij] = qq[0]*Y[0];
            q[16 + ij] = qq[1]*Y[1] + qq[2]*Y[2] + qq[3]*Y[3];
            q[25 + ij] = qq[4]*Y[4] + qq[5]*Y[5] + qq[6]*Y[6]
                       + qq[7]*Y[7] + qq[8]*Y[8];
        }
    }
    __syncthreads();

    // ---- Phase 2: weights[32x384] = h[32x64] @ W2[64x384] + b2 via WMMA f16.
    // 48 tiles (2 Mtiles x 24 Ntiles), 6 per wave, K=64 = 2 x (16x16x32) steps.
    #pragma unroll
    for (int t = 0; t < 6; ++t) {
        const int tile  = wave + t * 8;      // uniform per wave, all waves busy
        const int mtile = tile & 1;
        const int ntile = tile >> 1;
        const int half  = lane >> 4;
        const int mrow  = mtile * 16 + (lane & 15);   // A-matrix row for lane

        v8f acc = {};
        #pragma unroll
        for (int ks = 0; ks < 2; ++ks) {
            union { v16h v; uint4 q[2]; } A, B;
            // 16-bit A 16x32 fragment: elems 0..7 = K 8*half+0..7,
            //                          elems 8..15 = K 16+8*half+0..7
            const _Float16* hr = &h_lds[mrow][ks*32 + half*8];
            A.q[0] = *(const uint4*)(hr);
            A.q[1] = *(const uint4*)(hr + 16);
            // 16-bit B 32x16 fragment: lane&15 = N col, elems = K in same order as A
            const _Float16* br = W2h + (size_t)(ks*32)*NWEIGHT + ntile*16 + (lane & 15);
            #pragma unroll
            for (int s = 0; s < 16; ++s) {
                const int kb = (s < 8) ? (half*8 + s) : (16 + half*8 + s - 8);
                B.v[s] = br[(size_t)kb*NWEIGHT];
            }
            acc = __builtin_amdgcn_wmma_f32_16x16x32_f16(
                      false, A.v, false, B.v, (short)0, acc, false, false);
        }
        // D layout: lane&15 = N col, VGPR r = M row (+8 for upper half-wave)
        const int chan = ntile * 16 + (lane & 15);
        const float bias = b2[chan];
        #pragma unroll
        for (int r = 0; r < 8; ++r) {
            const int pr = mtile * 16 + half * 8 + r;
            w_lds[pr][chan] = acc[r] + bias;
        }
    }
    __syncthreads();

    // ---- Phase 3: assemble the 32x32 kernel block; lane = pair,
    // (row,col) wave-uniform -> contiguous 128B non-temporal stores.
    {
        const int pair = lane;
        const float* qp = qy_lds[pair];
        const float* wp = w_lds[pair];
        const size_t obase = (size_t)pairBase + pair;
        #pragma unroll 4
        for (int it = 0; it < 128; ++it) {
            const int rc  = wave * 128 + it;
            const int row = rc >> 5;
            const int col = rc & 31;
            float val;
            if (row < 8) {                       // l_out = 0 rows
                if (col < 8) {                   // block (0,0)
                    val = qp[0] * wp[row*8 + col];
                } else {                         // block (0,1)
                    const int v = (col - 8) / 3, j = (col - 8) % 3;
                    val = qp[1 + j] * wp[64 + row*8 + v];
                }
            } else {                             // l_out = 1 rows
                const int u = (row - 8) / 3, i = (row - 8) % 3;
                if (col < 8) {                   // block (1,0)
                    val = qp[4 + i] * wp[128 + u*8 + col];
                } else {                         // block (1,1), Js = {0,1,2}
                    const int v = (col - 8) / 3, j = (col - 8) % 3;
                    const int c  = 192 + (u*8 + v) * 3;
                    const int ij = i*3 + j;
                    val = qp[7  + ij] * wp[c]
                        + qp[16 + ij] * wp[c + 1]
                        + qp[25 + ij] * wp[c + 2];
                }
            }
            __builtin_nontemporal_store(val,
                out + (size_t)rc * (size_t)NPAIR_TOTAL + obase);
        }
    }
}

// ---------------------------------------------------------------------------
extern "C" void kernel_launch(void* const* d_in, const int* in_sizes, int n_in,
                              void* d_out, int out_size, void* d_ws, size_t ws_size,
                              hipStream_t stream) {
    const float* dmat = (const float*)d_in[0];
    const float* Q00  = (const float*)d_in[1];
    const float* Q01  = (const float*)d_in[2];
    const float* Q10  = (const float*)d_in[3];
    const float* Q11  = (const float*)d_in[4];
    const float* W1   = (const float*)d_in[5];
    const float* b1   = (const float*)d_in[6];
    // d_in[7] = W2 (f32) -- converted to f16 in workspace
    const float* W2   = (const float*)d_in[7];
    const float* b2   = (const float*)d_in[8];
    float* out = (float*)d_out;

    _Float16* W2h = (_Float16*)d_ws;   // 64*384*2 = 48 KB of workspace

    w2_to_f16<<<dim3((HIDDEN*NWEIGHT + 255) / 256), dim3(256), 0, stream>>>(W2, W2h);
    se3_fused<<<dim3(NPAIR_TOTAL / PAIRS), dim3(256), 0, stream>>>(
        dmat, Q00, Q01, Q10, Q11, W1, b1, W2h, b2, out);
}
